// NodeLayer1_81123342287185
// MI455X (gfx1250) — matrix operand order repaired and checked by reading, not compiled, not run on hardware
//
#include <hip/hip_runtime.h>
#include <hip/hip_bf16.h>
#include <math.h>

// ---------------------------------------------------------------------------
// MI455X (gfx1250): wave32, WMMA 16x16x32 bf16 (f32 accum).
// GEMM operands are pre-converted to bf16 (weights pre-transposed) so LDS
// staging is a pure byte copy -> Tensor Data Mover for the weight panel,
// per-lane async global->LDS for the A tile. No VGPR round-trip, no cvt in
// the hot loop. Workload is HBM-bound (~2GB moved, ~70 GFLOP), so GEMM cost
// must be near-zero and all scatter/gather kernels coalesced.
// ---------------------------------------------------------------------------

typedef __attribute__((ext_vector_type(16))) __bf16 v16bf;
typedef __attribute__((ext_vector_type(8)))  __bf16 v8bf;
typedef __attribute__((ext_vector_type(8)))  float  v8f;
typedef __attribute__((ext_vector_type(4)))  unsigned int v4u;
typedef __attribute__((ext_vector_type(4)))  int    v4i;
typedef __attribute__((ext_vector_type(8)))  int    v8i;

union V16U { v16bf v; v8bf h[2]; };

#if defined(__has_builtin)
#  if __has_builtin(__builtin_amdgcn_tensor_load_to_lds)
#    define HAS_TDM 1
#  else
#    define HAS_TDM 0
#  endif
#else
#  define HAS_TDM 0
#endif

__device__ inline void atomicMaxF32(float* addr, float v) {
  if (v >= 0.f) atomicMax((int*)addr, __float_as_int(v));
  else          atomicMin((unsigned int*)addr, __float_as_uint(v));
}

// per-lane async global->LDS 16B copy (ASYNCcnt path; inline asm = portable)
__device__ inline void async_ld16(const void* g, void* l) {
  asm volatile("global_load_async_to_lds_b128 %0, %1, off"
               :: "v"((unsigned)(uintptr_t)l),
                  "v"((unsigned long long)(uintptr_t)g)
               : "memory");
}
__device__ inline void wait_async0() {
  asm volatile("s_wait_asynccnt 0x0" ::: "memory");
}

#if HAS_TDM
// TDM: copy a contiguous [128][K] bf16 panel (2D tile, data_size=2B) to LDS.
// Descriptor bitfields per CDNA5 ISA.
__device__ inline void tdm_load_panel(const __bf16* g, void* lds, int K) {
  unsigned long long ga = (unsigned long long)(uintptr_t)g;
  unsigned lds_off = (unsigned)(uintptr_t)lds;          // generic low32 = LDS byte offset
  v4u g0 = { 1u,                                        // count=1
             lds_off,                                   // lds_addr [63:32]
             (unsigned)(ga & 0xFFFFFFFFu),              // global_addr lo
             (unsigned)(((ga >> 32) & 0x1FFFFFFu) | (2u << 30)) }; // addr hi | type=2
  unsigned d1s = (unsigned)(K * 128);                   // dim1 stride (elements)
  v8i g1 = { (int)(1u << 16),                           // data_size=1 (2 bytes)
             (int)(((unsigned)K & 0xFFFFu) << 16),      // tensor_dim0 lo16
             (int)(((unsigned)K >> 16) | (128u << 16)), // dim0 hi | tensor_dim1 lo16
             (int)(((unsigned)K & 0xFFFFu) << 16),      // dim1 hi(0) | tile_dim0=K
             (int)128,                                  // tile_dim1=128, tile_dim2=0
             (int)K,                                    // tensor_dim0_stride lo32
             (int)((d1s & 0xFFFFu) << 16),              // d0s hi(0) | d1s lo16
             (int)(d1s >> 16) };                        // d1s hi
  v4i z4 = { 0, 0, 0, 0 };
#if defined(__clang_major__) && __clang_major__ >= 23
  v8i z8 = { 0, 0, 0, 0, 0, 0, 0, 0 };
  __builtin_amdgcn_tensor_load_to_lds(g0, g1, z4, z4, z8, 0);
#else
  __builtin_amdgcn_tensor_load_to_lds(g0, g1, z4, z4, 0);
#endif
  __builtin_amdgcn_s_wait_tensorcnt(0);
}
#endif

// ---------------------------------------------------------------------------
// WMMA GEMM: out[M,128] = A[M,K](bf16) @ W[K,128] (+bias) (+resid)
// Wt = pre-transposed bf16 weight panel [128][K] (contiguous).
// Block = 256 threads = 8 waves; tile 32 rows x 128 cols.
// Fragment layouts per CDNA5 ISA 7.12.2 (16-bit A/B lane runs of 8/16).
// ---------------------------------------------------------------------------
__global__ void k_gemm_wmma_bf(const __bf16* __restrict__ A, int K,
                               const __bf16* __restrict__ Wt,
                               float* __restrict__ out,
                               const float* __restrict__ bias,
                               const float* __restrict__ resid,
                               long M)
{
  extern __shared__ __bf16 smem[];
  __bf16* sW = smem;                     // [128][K]
  __bf16* sA = smem + (size_t)128 * K;   // [32][K]
  const int tid = threadIdx.x;
  const long row0 = (long)blockIdx.x * 32;

#if HAS_TDM
  if (tid < 32) {                        // TDM is per-wave, EXEC-ignored
    tdm_load_panel(Wt, sW, K);
  }
#else
  for (int i = tid; i < (128 * K) / 4; i += 256)
    ((unsigned long long*)sW)[i] = ((const unsigned long long*)Wt)[i];
#endif

  // A tile: contiguous 32*K*2 bytes from row0 (clamped at M for tail block);
  // padding rows feed garbage into masked-out outputs only.
  {
    const size_t bytesA = (size_t)32 * K * 2;
    const size_t limit  = (size_t)M * K * 2 - 16;
    const size_t base   = (size_t)row0 * K * 2;
    const char*  gA = (const char*)A;
    char*        lA = (char*)sA;
    for (size_t off = (size_t)tid * 16; off < bytesA; off += 256 * 16) {
      size_t go = base + off;
      if (go > limit) go = limit;
      async_ld16(gA + go, lA + off);
    }
    wait_async0();
  }
  __syncthreads();

  const int wave = tid >> 5, lane = tid & 31;
  const int lr = lane & 15, hi = lane >> 4;
  const int n0 = wave * 16;

  v8f acc0 = {};
  v8f acc1 = {};
  for (int kb = 0; kb < K; kb += 32) {
    V16U a0, a1;
    a0.h[0] = *(const v8bf*)&sA[(size_t)lr * K + kb + hi * 8];
    a0.h[1] = *(const v8bf*)&sA[(size_t)lr * K + kb + 16 + hi * 8];
    a1.h[0] = *(const v8bf*)&sA[(size_t)(16 + lr) * K + kb + hi * 8];
    a1.h[1] = *(const v8bf*)&sA[(size_t)(16 + lr) * K + kb + 16 + hi * 8];
    V16U bu;
    bu.h[0] = *(const v8bf*)&sW[(size_t)(n0 + lr) * K + kb + hi * 8];
    bu.h[1] = *(const v8bf*)&sW[(size_t)(n0 + lr) * K + kb + 16 + hi * 8];
    v16bf b = bu.v;
    acc0 = __builtin_amdgcn_wmma_f32_16x16x32_bf16(false, a0.v, false, b,
                                                   (short)0, acc0, false, false);
    acc1 = __builtin_amdgcn_wmma_f32_16x16x32_bf16(false, a1.v, false, b,
                                                   (short)0, acc1, false, false);
  }

  const int gn = n0 + lr;
  const float bv = bias ? bias[gn] : 0.f;
#pragma unroll
  for (int r = 0; r < 8; ++r) {
    long m0 = row0 + r + hi * 8;
    long m1 = row0 + 16 + r + hi * 8;
    if (m0 < M) {
      float v = acc0[r] + bv;
      if (resid) v += resid[m0 * 128 + gn];
      out[m0 * 128 + gn] = v;
    }
    if (m1 < M) {
      float v = acc1[r] + bv;
      if (resid) v += resid[m1 * 128 + gn];
      out[m1 * 128 + gn] = v;
    }
  }
}

// ------------------------------ prep kernels -------------------------------

// Wt[n*K+k] = (bf16)W[k*128+n]  (coalesced read; tiny panels, stay hot in L2)
__global__ void k_prep_wt(const float* __restrict__ W, __bf16* __restrict__ Wt, int K) {
  long gid = (long)blockIdx.x * blockDim.x + threadIdx.x;
  if (gid >= (long)128 * K) return;
  int k = (int)(gid >> 7), n = (int)(gid & 127);
  Wt[(size_t)n * K + k] = (__bf16)W[gid];
}

__global__ void k_cast_bf16(const float* __restrict__ x, __bf16* __restrict__ y, long n4) {
  long gid = (long)blockIdx.x * blockDim.x + threadIdx.x;
  if (gid >= n4) return;
  const float4 v = ((const float4*)x)[gid];
  union { __bf16 b[4]; unsigned long long u; } t;
  t.b[0] = (__bf16)v.x; t.b[1] = (__bf16)v.y;
  t.b[2] = (__bf16)v.z; t.b[3] = (__bf16)v.w;
  ((unsigned long long*)y)[gid] = t.u;
}

// --------------------------- graph / edge kernels --------------------------

__global__ void k_fillf(float* __restrict__ p, long n, float v) {
  long gid = (long)blockIdx.x * blockDim.x + threadIdx.x;
  if (gid < n) p[gid] = v;
}

__global__ void k_init_x4(float* __restrict__ x4, long N,
                          const float* __restrict__ b1, const float* __restrict__ b2,
                          const float* __restrict__ b3, const float* __restrict__ b4) {
  long gid = (long)blockIdx.x * blockDim.x + threadIdx.x;
  if (gid >= N * 512) return;
  int col = (int)(gid & 511);
  int c = col >> 7, cc = col & 127;
  const float* bp = (c == 0) ? b1 : (c == 1) ? b2 : (c == 2) ? b3 : b4;
  x4[gid] = bp[cc];
}

__global__ void k_edge_logits(long Ec, const int* __restrict__ src, const int* __restrict__ dst,
                              const float* __restrict__ xl, const float* __restrict__ xr,
                              const float* __restrict__ me, const float* __restrict__ att,
                              float* __restrict__ logits, float* __restrict__ nmax) {
  long gid = (long)blockIdx.x * blockDim.x + threadIdx.x;
  if (gid >= Ec * 16) return;
  long e = gid >> 4; int h = (int)(gid & 15);
  long s = src[e], d = dst[e];
  const float* pa = xl + s * 128 + h * 8;
  const float* pb = xr + d * 128 + h * 8;
  float acc = 0.f;
#pragma unroll
  for (int j = 0; j < 8; ++j) {
    float m = pa[j] + pb[j];
    if (me) m += me[e * 128 + h * 8 + j];
    m = (m > 0.f) ? m : 0.2f * m;
    acc += m * att[h * 8 + j];
  }
  logits[gid] = acc;
  atomicMaxF32(&nmax[d * 16 + h], acc);
}

__global__ void k_edge_expsum(long Ec, const int* __restrict__ dst,
                              float* __restrict__ logits, const float* __restrict__ nmax,
                              float* __restrict__ nsum) {
  long gid = (long)blockIdx.x * blockDim.x + threadIdx.x;
  if (gid >= Ec * 16) return;
  long e = gid >> 4; int h = (int)(gid & 15);
  long d = dst[e];
  float v = __expf(logits[gid] - nmax[d * 16 + h]);
  logits[gid] = v;
  atomicAdd(&nsum[d * 16 + h], v);
}

__global__ void k_edge_aggregate(long Ec, const int* __restrict__ src, const int* __restrict__ dst,
                                 const float* __restrict__ eexp, const float* __restrict__ nsum,
                                 const float* __restrict__ xl, float* __restrict__ x4, int conv) {
  long gid = (long)blockIdx.x * blockDim.x + threadIdx.x;
  if (gid >= Ec * 128) return;
  long e = gid >> 7; int t = (int)(gid & 127); int h = t >> 3;
  long s = src[e], d = dst[e];
  float alpha = eexp[e * 16 + h] / (nsum[d * 16 + h] + 1e-16f);
  atomicAdd(&x4[d * 512 + conv * 128 + t], alpha * xl[s * 128 + t]);
}

__global__ void k_pair_ln(long Ec, const int* __restrict__ src, const int* __restrict__ dst,
                          const float* __restrict__ xf, const float* __restrict__ g,
                          const float* __restrict__ beta, __bf16* __restrict__ pair) {
  int wave = threadIdx.x >> 5, lane = threadIdx.x & 31;
  long e = (long)blockIdx.x * 8 + wave;
  if (e >= Ec) return;
  long s = src[e], d = dst[e];
  int j0 = lane * 8;
  float v[8];
#pragma unroll
  for (int j = 0; j < 8; ++j) {
    int col = j0 + j;
    v[j] = (col < 128) ? xf[s * 128 + col] : xf[d * 128 + (col - 128)];
  }
  float sum = 0.f, sq = 0.f;
#pragma unroll
  for (int j = 0; j < 8; ++j) { sum += v[j]; sq += v[j] * v[j]; }
#pragma unroll
  for (int m = 16; m >= 1; m >>= 1) {
    sum += __shfl_xor(sum, m);
    sq  += __shfl_xor(sq,  m);
  }
  float mean = sum * (1.f / 256.f);
  float var  = sq * (1.f / 256.f) - mean * mean;
  float rn   = rsqrtf(var + 1e-5f);
#pragma unroll
  for (int j = 0; j < 8; ++j) {
    int col = j0 + j;
    float h = (v[j] - mean) * rn * g[col] + beta[col];
    h = (h > 0.f) ? h : 0.f;
    pair[e * 256 + col] = (__bf16)h;
  }
}

__global__ void k_gather_dual(long NO, long E, const int* __restrict__ idx,
                              const float* __restrict__ ean, float* __restrict__ dual) {
  long gid = (long)blockIdx.x * blockDim.x + threadIdx.x;
  if (gid >= NO * 128) return;
  long r = gid >> 7; int c = (int)(gid & 127);
  long id = (long)(unsigned)idx[r];
  dual[gid] = (id < E) ? ean[id * 128 + c] : 1.0f;
}

// ------------------------------- launcher ----------------------------------

extern "C" void kernel_launch(void* const* d_in, const int* in_sizes, int n_in,
                              void* d_out, int out_size, void* d_ws, size_t ws_size,
                              hipStream_t stream) {
  const float* x         = (const float*)d_in[0];
  const int*   ei[4]     = {(const int*)d_in[1], (const int*)d_in[2],
                            (const int*)d_in[3], (const int*)d_in[4]};
  const float* dual_attr = (const float*)d_in[5];
  const float* edge_attr = (const float*)d_in[6];
  const int*   oemap     = (const int*)d_in[7];
  const float *Wl[4], *Wr[4], *att[4], *bias[4];
  for (int c = 0; c < 4; ++c) {
    Wl[c]   = (const float*)d_in[8  + 4 * c];
    Wr[c]   = (const float*)d_in[9  + 4 * c];
    att[c]  = (const float*)d_in[10 + 4 * c];
    bias[c] = (const float*)d_in[11 + 4 * c];
  }
  const float* We1    = (const float*)d_in[24];
  const float* fuse_W = (const float*)d_in[25];
  const float* fuse_b = (const float*)d_in[26];
  const float* ln_g   = (const float*)d_in[27];
  const float* ln_b   = (const float*)d_in[28];
  const float* mlp_W  = (const float*)d_in[29];
  const float* mlp_b  = (const float*)d_in[30];

  const long N  = in_sizes[0] / 128;
  const long Ec[4] = { in_sizes[1] / 2, in_sizes[2] / 2,
                       in_sizes[3] / 2, in_sizes[4] / 2 };
  const long E  = Ec[0];
  const long NO = in_sizes[7];

  // ---- workspace layout ----
  float* p = (float*)d_ws;
  float *xl[4], *xr[4];
  for (int c = 0; c < 4; ++c) { xl[c] = p; p += N * 128; xr[c] = p; p += N * 128; }
  float* me   = p; p += E * 128;          // reused as bf16 pair buffer (same bytes)
  float* elog = p; p += E * 16;
  float* nmax = p; p += N * 16;
  float* nsum = p; p += N * 16;
  float* x4   = p; p += N * 512;
  __bf16* bp2 = (__bf16*)p;
  __bf16* x_bf    = bp2; bp2 += N * 128;
  __bf16* x4_bf   = bp2; bp2 += N * 512;
  __bf16* dual_bf = bp2; bp2 += E * 128;
  __bf16* wt_l[4], *wt_r[4];
  for (int c = 0; c < 4; ++c) { wt_l[c] = bp2; bp2 += 128 * 128; wt_r[c] = bp2; bp2 += 128 * 128; }
  __bf16* wt_we   = bp2; bp2 += 128 * 128;
  __bf16* wt_fuse = bp2; bp2 += 128 * 512;
  __bf16* wt_mlp  = bp2; bp2 += 128 * 256;
  __bf16* pairbf  = (__bf16*)me;          // E*256 bf16 == E*128 f32 bytes

  float* xf   = (float*)d_out;            // [N,128]
  float* ean  = xf + N * 128;             // [E,128]
  float* dual = ean + E * 128;            // [NO,128]

  auto cdiv = [](long a, long b) { return (unsigned)((a + b - 1) / b); };

  // 0) prep: transposed bf16 weight panels + bf16 casts of A matrices
  for (int c = 0; c < 4; ++c) {
    k_prep_wt<<<cdiv(128 * 128, 256), 256, 0, stream>>>(Wl[c], wt_l[c], 128);
    k_prep_wt<<<cdiv(128 * 128, 256), 256, 0, stream>>>(Wr[c], wt_r[c], 128);
  }
  k_prep_wt<<<cdiv(128 * 128, 256), 256, 0, stream>>>(We1,    wt_we,   128);
  k_prep_wt<<<cdiv(128 * 512, 256), 256, 0, stream>>>(fuse_W, wt_fuse, 512);
  k_prep_wt<<<cdiv(128 * 256, 256), 256, 0, stream>>>(mlp_W,  wt_mlp,  256);
  k_cast_bf16<<<cdiv(N * 32, 256), 256, 0, stream>>>(x, x_bf, N * 32);
  k_cast_bf16<<<cdiv(E * 32, 256), 256, 0, stream>>>(dual_attr, dual_bf, E * 32);

  // 1) node projections (K=128, 40KB LDS)
  for (int c = 0; c < 4; ++c) {
    k_gemm_wmma_bf<<<cdiv(N, 32), 256, 320 * 128, stream>>>(
        x_bf, 128, wt_l[c], xl[c], nullptr, nullptr, N);
    k_gemm_wmma_bf<<<cdiv(N, 32), 256, 320 * 128, stream>>>(
        x_bf, 128, wt_r[c], xr[c], nullptr, nullptr, N);
  }
  // 2) edge feature projection (conv1)
  k_gemm_wmma_bf<<<cdiv(E, 32), 256, 320 * 128, stream>>>(
      dual_bf, 128, wt_we, me, nullptr, nullptr, E);

  // 3) fused accumulator pre-loaded with biases
  k_init_x4<<<cdiv(N * 512, 256), 256, 0, stream>>>(x4, N, bias[0], bias[1], bias[2], bias[3]);

  // 4) per-conv edge softmax + aggregation
  for (int c = 0; c < 4; ++c) {
    k_fillf<<<cdiv(N * 16, 256), 256, 0, stream>>>(nmax, N * 16, -INFINITY);
    k_fillf<<<cdiv(N * 16, 256), 256, 0, stream>>>(nsum, N * 16, 0.f);
    const int* s = ei[c];
    const int* d = ei[c] + Ec[c];
    const float* mep = (c == 0) ? me : nullptr;
    k_edge_logits<<<cdiv(Ec[c] * 16, 256), 256, 0, stream>>>(
        Ec[c], s, d, xl[c], xr[c], mep, att[c], elog, nmax);
    k_edge_expsum<<<cdiv(Ec[c] * 16, 256), 256, 0, stream>>>(
        Ec[c], d, elog, nmax, nsum);
    k_edge_aggregate<<<cdiv(Ec[c] * 128, 256), 256, 0, stream>>>(
        Ec[c], s, d, elog, nsum, xl[c], x4, c);
  }

  // 5) fuse GEMM (K=512, 160KB LDS)
  k_cast_bf16<<<cdiv(N * 128, 256), 256, 0, stream>>>(x4, x4_bf, N * 128);
  k_gemm_wmma_bf<<<cdiv(N, 32), 256, 320 * 512, stream>>>(
      x4_bf, 512, wt_fuse, xf, fuse_b, nullptr, N);

  // 6) per-edge layernorm+relu -> bf16 pair rows (me region now free)
  k_pair_ln<<<cdiv(E, 8), 256, 0, stream>>>(E, ei[0], ei[0] + E, xf, ln_g, ln_b, pairbf);

  // 7) edge MLP + residual (K=256, 80KB LDS)
  k_gemm_wmma_bf<<<cdiv(E, 32), 256, 320 * 256, stream>>>(
      pairbf, 256, wt_mlp, ean, mlp_b, edge_attr, E);

  // 8) dual gather with ones padding row
  k_gather_dual<<<cdiv(NO * 128, 256), 256, 0, stream>>>(NO, E, oemap, ean, dual);
}
